// BiLSTMCRF_25039659336327
// MI455X (gfx1250) — matrix operand order repaired and checked by reading, not compiled, not run on hardware
//
#include <hip/hip_runtime.h>
#include <math.h>

// ---------- types for CDNA5 WMMA ----------
typedef __bf16 v16bf __attribute__((ext_vector_type(16)));
typedef float  v8f   __attribute__((ext_vector_type(8)));
typedef unsigned int v4u __attribute__((ext_vector_type(4)));

union FragB16 { v4u u[2]; v16bf v; };   // 16 bf16 = 32B = 8 VGPRs
union FragC   { float f[8]; v8f v; };   // 16x16 f32 C/D tile slice

__device__ __forceinline__ unsigned short f2bf(float x) {
    unsigned int b = __float_as_uint(x);
    unsigned int r = b + 0x7FFFu + ((b >> 16) & 1u);   // round-to-nearest-even
    return (unsigned short)(r >> 16);
}

// ---------- f32 -> bf16 weight conversion ----------
__global__ void k_f32_to_bf16(const float* __restrict__ in,
                              unsigned short* __restrict__ out, int n) {
    int i = blockIdx.x * 256 + threadIdx.x;
    if (i < n) out[i] = f2bf(in[i]);
}

// ---------- embedding gather: x_bf[t][b][e] = bf16(embed[sent[b][t]][e]) ----------
__global__ __launch_bounds__(64) void k_embed(const int* __restrict__ sent,
                                              const float* __restrict__ embed,
                                              unsigned short* __restrict__ xbf,
                                              int B, int T, int E) {
    int tb = blockIdx.x;          // t*B + b
    int t = tb / B, b = tb % B;
    int tok = sent[b * T + t];
    const float4* src = (const float4*)(embed + (size_t)tok * E);
    ushort4* dst = (ushort4*)(xbf + (size_t)tb * E);
    for (int e = threadIdx.x; e < (E >> 2); e += blockDim.x) {
        float4 v = src[e];
        ushort4 o;
        o.x = f2bf(v.x); o.y = f2bf(v.y); o.z = f2bf(v.z); o.w = f2bf(v.w);
        dst[e] = o;
    }
}

// ---------- WMMA GEMM: C[M,N] = A[M,K](bf16) * W[N,K](bf16)^T + bias ----------
// Compile-time K, N, NT (n-tiles per wave). block = 256 threads (8 waves):
// wave w -> mtile = blockIdx.y*8+w (16 rows of A/C), NT n-tiles at blockIdx.x*16*NT.
// All loads are immediate-offset b128 from two per-thread byte bases.
template <int K, int N, int NT>
__global__ __launch_bounds__(256) void k_gemm(
    const unsigned short* __restrict__ A,   // M x K bf16 (row major)
    const unsigned short* __restrict__ W,   // N x K bf16 (row major)
    const float* __restrict__ bias,         // N
    float* __restrict__ C)                  // M x N f32
{
    const int lane = threadIdx.x & 31;
    const int wave = threadIdx.x >> 5;
    const int half = lane >> 4;
    const int l16  = lane & 15;
    const int mtile = blockIdx.y * 8 + wave;
    const int m = mtile * 16 + l16;         // A row for this lane
    const int nbase = blockIdx.x * (16 * NT);

    // byte bases (element = 2 bytes)
    const char* Ab = (const char*)(A + (size_t)m * K + half * 8);
    const char* Wb = (const char*)(W + (size_t)(nbase + l16) * K + half * 8);

    FragC acc[NT];
#pragma unroll
    for (int j = 0; j < NT; j++)
#pragma unroll
        for (int r = 0; r < 8; r++) acc[j].f[r] = 0.0f;

#pragma unroll
    for (int ks = 0; ks < (K >> 5); ks++) {
        FragB16 a;
        a.u[0] = *(const v4u*)(Ab + ks * 64);        // K = kb + half*8 + [0..7]
        a.u[1] = *(const v4u*)(Ab + ks * 64 + 32);   // K = kb + 16 + half*8 + [0..7]
#pragma unroll
        for (int j = 0; j < NT; j++) {
            FragB16 b;
            b.u[0] = *(const v4u*)(Wb + j * (32 * K) + ks * 64);
            b.u[1] = *(const v4u*)(Wb + j * (32 * K) + ks * 64 + 32);
            acc[j].v = __builtin_amdgcn_wmma_f32_16x16x32_bf16(
                false, a.v, false, b.v, (short)0, acc[j].v, false, false);
        }
    }

    float* Cb = C + (size_t)(mtile * 16 + half * 8) * N + nbase + l16;
#pragma unroll
    for (int j = 0; j < NT; j++) {
        const float bs = bias[nbase + j * 16 + l16];
#pragma unroll
        for (int r = 0; r < 8; r++) Cb[(size_t)r * N + j * 16] = acc[j].f[r] + bs;
    }
}

// ---------- sequential BiLSTM layer (H=256, B=32 hardcoded) ----------
// grid = 2 (dir), block = 1024 (32 waves). wave w: mtile = w&1, unit-tile ut = w>>1.
// Each wave computes gate rows {g*H + ut*16 .. +16} for all 4 gates so i/f/g/o for
// its units stay in-wave; cell state c in registers; carried h in LDS (bf16, padded).
// Whh is streamed from L2 every step (512KB/dir): an opaque zero offset (asm on an
// SGPR value, NOT on the pointer) defeats LICM so the 64 B-fragment loads stay inside
// the time loop, while keeping the pointer's global address space known -> the loads
// stay global_load_b128 (not flat_load, which would also burn DScnt / the LDS path).
#define LSTM_HPAD 264   // padded row stride (bf16 elems) for LDS bank spread

__global__ __launch_bounds__(1024) void k_lstm(
    const float* __restrict__ Gpre_f,        // (T,B,4H) precomputed x@Wih^T + b, fwd
    const float* __restrict__ Gpre_b,        // backward dir
    const unsigned short* __restrict__ Whh_f,// (4H,H) bf16
    const unsigned short* __restrict__ Whh_b,
    const int* __restrict__ lengths,         // B
    unsigned short* __restrict__ xcat,       // (T,B,2H) bf16 out: [hf | hb]
    int T)
{
    constexpr int H = 256, B = 32, H4n = 1024;
    __shared__ unsigned short h_bf[32 * LSTM_HPAD];

    const int dir = blockIdx.x;
    const float* Gpre = dir ? Gpre_b : Gpre_f;
    const unsigned short* Whh = dir ? Whh_b : Whh_f;

    const int lane = threadIdx.x & 31;
    const int wave = threadIdx.x >> 5;
    const int half = lane >> 4;
    const int l16  = lane & 15;
    const int mtile = wave & 1;
    const int ut    = wave >> 1;             // 0..15
    const int u     = ut * 16 + l16;         // hidden unit 0..255
    const int mrow  = mtile * 16 + half * 8; // batch row base (+r)
    const int am    = mtile * 16 + l16;      // A-matrix row for this lane

    // byte bases (computed once)
    const char* Wbase = (const char*)(Whh + (size_t)u * H + half * 8);
    const char* Hbase = (const char*)(h_bf + (size_t)am * LSTM_HPAD + half * 8);
    unsigned short* hw = h_bf + (size_t)mrow * LSTM_HPAD + u;         // + r*LSTM_HPAD
    unsigned short* xw = xcat + (size_t)mrow * (2 * H) + dir * H + u; // + r*512

    for (int i = threadIdx.x; i < 32 * LSTM_HPAD; i += blockDim.x) h_bf[i] = 0;

    int len_r[8];
    float c_r[8];
#pragma unroll
    for (int r = 0; r < 8; r++) { len_r[r] = lengths[mrow + r]; c_r[r] = 0.0f; }
    __syncthreads();

    size_t zoff = 0;   // always 0 at runtime; opaque to the optimizer per iteration
    for (int s = 0; s < T; s++) {
        const int t = dir ? (T - 1 - s) : s;

        // new opaque value each timestep -> Whh loads cannot be hoisted/spilled
        asm volatile("" : "+s"(zoff));
        const char* Wstep = Wbase + zoff;

        // prefetch gate pre-activations (independent of h) -> overlap with WMMA
        const float* Gt = Gpre + (size_t)t * B * H4n + (size_t)mrow * H4n + u;
        float gp[4][8];
#pragma unroll
        for (int g = 0; g < 4; g++)
#pragma unroll
            for (int r = 0; r < 8; r++) gp[g][r] = Gt[(size_t)r * H4n + g * H];

        FragC acc[4];
#pragma unroll
        for (int g = 0; g < 4; g++)
#pragma unroll
            for (int r = 0; r < 8; r++) acc[g].f[r] = 0.0f;

#pragma unroll
        for (int ks = 0; ks < (H >> 5); ks++) {   // 8 k-steps
            FragB16 a;
            a.u[0] = *(const v4u*)(Hbase + ks * 64);
            a.u[1] = *(const v4u*)(Hbase + ks * 64 + 32);
#pragma unroll
            for (int g = 0; g < 4; g++) {
                FragB16 b;
                b.u[0] = *(const v4u*)(Wstep + g * (2 * H * H) + ks * 64);
                b.u[1] = *(const v4u*)(Wstep + g * (2 * H * H) + ks * 64 + 32);
                acc[g].v = __builtin_amdgcn_wmma_f32_16x16x32_bf16(
                    false, a.v, false, b.v, (short)0, acc[g].v, false, false);
            }
        }
        __syncthreads();   // all waves finished reading h_bf for this step

        unsigned short* xwt = xw + (size_t)t * B * (2 * H);
#pragma unroll
        for (int r = 0; r < 8; r++) {
            float gi = gp[0][r] + acc[0].f[r];
            float gf = gp[1][r] + acc[1].f[r];
            float gg = gp[2][r] + acc[2].f[r];
            float go = gp[3][r] + acc[3].f[r];
            float ii = 1.0f / (1.0f + __expf(-gi));
            float ff = 1.0f / (1.0f + __expf(-gf));
            float tg = tanhf(gg);
            float oo = 1.0f / (1.0f + __expf(-go));
            unsigned short hb_out = 0;
            if (t < len_r[r]) {
                float c2 = ff * c_r[r] + ii * tg;
                float h2 = oo * tanhf(c2);
                c_r[r] = c2;
                hb_out = f2bf(h2);
                hw[r * LSTM_HPAD] = hb_out;       // carry (kept when masked)
            }
            xwt[r * (2 * H)] = hb_out;            // output: h2 where mask else 0
        }
        __syncthreads();   // h_bf updated; safe to read next step
    }
}

// ---------- CRF numerator (gold path score), one thread per batch ----------
__global__ void k_crf_num(const float* __restrict__ emit_tm, const int* __restrict__ tags,
                          const int* __restrict__ lengths, const float* __restrict__ start_t,
                          const float* __restrict__ end_t, const float* __restrict__ trans,
                          float* __restrict__ num, int T, int B, int K)
{
    int b = threadIdx.x;
    if (b >= B) return;
    int len = lengths[b];
    int prev = tags[b * T];
    float s = start_t[prev] + emit_tm[(size_t)b * K + prev];
    for (int t = 1; t < T; t++) {
        int tg = tags[b * T + t];
        float mt = (t < len) ? 1.0f : 0.0f;
        s += (trans[prev * K + tg] + emit_tm[((size_t)t * B + b) * K + tg]) * mt;
        prev = tg;
    }
    s += end_t[tags[b * T + (len - 1)]];
    num[b] = s;
}

// ---------- CRF partition function (forward algorithm), one block per batch ----------
__global__ __launch_bounds__(64) void k_crf_den(
    const float* __restrict__ emit_tm, const int* __restrict__ lengths,
    const float* __restrict__ start_t, const float* __restrict__ end_t,
    const float* __restrict__ trans, float* __restrict__ den, int T, int B, int K)
{
    __shared__ float sc[64];
    __shared__ float ns[64];
    const int b = blockIdx.x;
    const int j = threadIdx.x;
    const int len = lengths[b];
    if (j < K) sc[j] = start_t[j] + emit_tm[(size_t)b * K + j];
    __syncthreads();
    for (int t = 1; t < T; t++) {
        if (t < len) {                      // uniform over block
            if (j < K) {
                float mx = -1e30f;
                for (int i = 0; i < K; i++) mx = fmaxf(mx, sc[i] + trans[i * K + j]);
                float ss = 0.0f;
                for (int i = 0; i < K; i++) ss += __expf(sc[i] + trans[i * K + j] - mx);
                ns[j] = emit_tm[((size_t)t * B + b) * K + j] + mx + __logf(ss);
            }
            __syncthreads();
            if (j < K) sc[j] = ns[j];
            __syncthreads();
        }
    }
    if (j == 0) {
        float mx = -1e30f;
        for (int i = 0; i < K; i++) mx = fmaxf(mx, sc[i] + end_t[i]);
        float ss = 0.0f;
        for (int i = 0; i < K; i++) ss += __expf(sc[i] + end_t[i] - mx);
        den[b] = mx + __logf(ss);
    }
}

// ---------- finalize: emit (T,B,K)->(B,T,K), loss, mask ----------
__global__ void k_finalize(const float* __restrict__ num, const float* __restrict__ den,
                           const float* __restrict__ emit_tm, const int* __restrict__ lengths,
                           float* __restrict__ out, int T, int B, int K)
{
    const size_t total = (size_t)B * T * K;
    for (size_t i = blockIdx.x * (size_t)blockDim.x + threadIdx.x; i < total;
         i += (size_t)gridDim.x * blockDim.x) {
        int k = (int)(i % K);
        size_t bt = i / K;
        int t = (int)(bt % T);
        int b = (int)(bt / T);
        out[i] = emit_tm[((size_t)t * B + b) * K + k];
    }
    for (size_t i = blockIdx.x * (size_t)blockDim.x + threadIdx.x; i < (size_t)B * T;
         i += (size_t)gridDim.x * blockDim.x) {
        int t = (int)(i % T);
        int b = (int)(i / T);
        out[total + 1 + i] = (t < lengths[b]) ? 1.0f : 0.0f;
    }
    if (blockIdx.x == 0 && threadIdx.x == 0) {
        float s = 0.0f;
        for (int b = 0; b < B; b++) s += num[b] - den[b];
        out[total] = s / (float)B;
    }
}

// ---------- host launcher ----------
extern "C" void kernel_launch(void* const* d_in, const int* in_sizes, int n_in,
                              void* d_out, int out_size, void* d_ws, size_t ws_size,
                              hipStream_t stream)
{
    (void)in_sizes; (void)n_in; (void)out_size; (void)ws_size;
    const int B = 32, T = 256, E = 256, H = 256, K = 48;
    const int TB = T * B, H4 = 4 * H, H2 = 2 * H;

    const int*   sent    = (const int*)d_in[0];
    const int*   lengths = (const int*)d_in[1];
    const int*   tags    = (const int*)d_in[2];
    const float* embed   = (const float*)d_in[3];
    const float* Wih[4]  = {(const float*)d_in[4],  (const float*)d_in[7],
                            (const float*)d_in[10], (const float*)d_in[13]};
    const float* Whh[4]  = {(const float*)d_in[5],  (const float*)d_in[8],
                            (const float*)d_in[11], (const float*)d_in[14]};
    const float* bvec[4] = {(const float*)d_in[6],  (const float*)d_in[9],
                            (const float*)d_in[12], (const float*)d_in[15]};
    const float* Wout    = (const float*)d_in[16];
    const float* bout    = (const float*)d_in[17];
    const float* start_t = (const float*)d_in[18];
    const float* end_t   = (const float*)d_in[19];
    const float* trans   = (const float*)d_in[20];

    char* ws = (char*)d_ws;
    size_t off = 0;
    auto alloc = [&](size_t bytes) -> char* {
        char* p = ws + off; off += (bytes + 255) & ~(size_t)255; return p;
    };
    unsigned short* x_bf   = (unsigned short*)alloc((size_t)TB * E * 2);
    unsigned short* xcat1  = (unsigned short*)alloc((size_t)TB * H2 * 2);
    unsigned short* xcat2  = (unsigned short*)alloc((size_t)TB * H2 * 2);
    float* Gpre_f          = (float*)alloc((size_t)TB * H4 * 4);
    float* Gpre_b          = (float*)alloc((size_t)TB * H4 * 4);
    float* emit_tm         = (float*)alloc((size_t)TB * K * 4);
    const int WihK[4] = {E, E, H2, H2};
    unsigned short* Wih_bf[4];
    unsigned short* Whh_bf[4];
    for (int i = 0; i < 4; i++) {
        Wih_bf[i] = (unsigned short*)alloc((size_t)H4 * WihK[i] * 2);
        Whh_bf[i] = (unsigned short*)alloc((size_t)H4 * H * 2);
    }
    unsigned short* Wout_bf = (unsigned short*)alloc((size_t)K * H2 * 2);
    float* numb = (float*)alloc(B * 4);
    float* denb = (float*)alloc(B * 4);

    auto cvt = [&](const float* src, unsigned short* dst, int n) {
        k_f32_to_bf16<<<(n + 255) / 256, 256, 0, stream>>>(src, dst, n);
    };
    for (int i = 0; i < 4; i++) {
        cvt(Wih[i], Wih_bf[i], H4 * WihK[i]);
        cvt(Whh[i], Whh_bf[i], H4 * H);
    }
    cvt(Wout, Wout_bf, K * H2);

    k_embed<<<TB, 64, 0, stream>>>(sent, embed, x_bf, B, T, E);

    // ---- layer 0: input projections (WMMA GEMMs), then recurrence ----
    {
        dim3 grid(H4 / 64, TB / 128);   // (16, 64)
        k_gemm<256, 1024, 4><<<grid, 256, 0, stream>>>(x_bf, Wih_bf[0], bvec[0], Gpre_f);
        k_gemm<256, 1024, 4><<<grid, 256, 0, stream>>>(x_bf, Wih_bf[1], bvec[1], Gpre_b);
        k_lstm<<<2, 1024, 0, stream>>>(Gpre_f, Gpre_b, Whh_bf[0], Whh_bf[1],
                                       lengths, xcat1, T);
    }
    // ---- layer 1 ----
    {
        dim3 grid(H4 / 64, TB / 128);
        k_gemm<512, 1024, 4><<<grid, 256, 0, stream>>>(xcat1, Wih_bf[2], bvec[2], Gpre_f);
        k_gemm<512, 1024, 4><<<grid, 256, 0, stream>>>(xcat1, Wih_bf[3], bvec[3], Gpre_b);
        k_lstm<<<2, 1024, 0, stream>>>(Gpre_f, Gpre_b, Whh_bf[2], Whh_bf[3],
                                       lengths, xcat2, T);
    }
    // ---- emission scores: (TB x 512) @ (512 x 48) ----
    {
        dim3 grid(1, TB / 128);         // 3 n-tiles cover N=48 exactly
        k_gemm<512, 48, 3><<<grid, 256, 0, stream>>>(xcat2, Wout_bf, bout, emit_tm);
    }
    // ---- CRF ----
    k_crf_num<<<1, 32, 0, stream>>>(emit_tm, tags, lengths, start_t, end_t, trans,
                                    numb, T, B, K);
    k_crf_den<<<B, 64, 0, stream>>>(emit_tm, lengths, start_t, end_t, trans,
                                    denb, T, B, K);
    k_finalize<<<512, 256, 0, stream>>>(numb, denb, emit_tm, lengths,
                                        (float*)d_out, T, B, K);
}
